// QINLike_71588514889822
// MI455X (gfx1250) — matrix operand order repaired and checked by reading, not compiled, not run on hardware
//
#include <hip/hip_runtime.h>
#include <hip/hip_bf16.h>
#include <math.h>

// ---------------------------------------------------------------------------
// QIN-like recommender, fused for MI455X (gfx1250, wave32, WMMA + TDM).
//   B=1024 L=1024 NCAT=20 NNUM=10 E=32 A=128 H=256 QH=256 TOPK=30
// Algebraic fusion removes the B*L*E*A k/v projections:
//   scores[b,l] = seq_e[b,l] . (Wk @ q[b]) / sqrt(A)
//   interest[b] = (sum_k w_k seq_e[b,idx_k]) @ (Wv@Wo) + bo
// Dense tail (MLP + 2 quadratic blocks) runs as WMMA f16->f32 tile GEMMs,
// 16 batch rows per workgroup. Weights are pre-packed once into the exact
// wave32 WMMA B-fragment layout (f16, L2-resident), so each fragment load is
// one contiguous 32B per-lane load. Fused tile staged to LDS by the Tensor
// Data Mover (tensor_load_to_lds + s_wait_tensorcnt).
// ---------------------------------------------------------------------------

#define BB   1024
#define LL   1024
#define NCAT 20
#define NNUM 10
#define EE   32
#define AA   128
#define HH   256
#define TOPK 30

typedef __attribute__((ext_vector_type(16))) _Float16 v16h;
typedef __attribute__((ext_vector_type(8)))  float    v8f;
typedef uint32_t u32x4 __attribute__((ext_vector_type(4)));
typedef uint32_t u32x8 __attribute__((ext_vector_type(8)));

// ---- wave32 WMMA fragment builders (16-bit A 16x32, B 32x16, C/D f32 16x16)

__device__ __forceinline__ v16h load_a_frag(const _Float16* S, int lda,
                                            int kbase, int lane) {
  // A 16x32 f16: lanes 0-15 row m=lane, V0..3 = K{0..7}, V4..7 = K{16..23};
  // lanes 16-31 row m=lane-16, K ranges +8.
  v16h a;
  const int grp = lane >> 4;
  const int m   = lane & 15;
#pragma unroll
  for (int v = 0; v < 8; ++v) {
#pragma unroll
    for (int j = 0; j < 2; ++j) {
      int kk = (v < 4) ? (v * 2 + j + grp * 8) : (16 + (v - 4) * 2 + j + grp * 8);
      a[v * 2 + j] = S[m * lda + kbase + kk];
    }
  }
  return a;
}

// Packed B fragments: tile (kt,nt) of a [K,256] matrix stored as 32 lanes x
// 16 halves contiguously -> one aligned 32B load per lane.
__device__ __forceinline__ v16h load_b_packed(const _Float16* __restrict__ P,
                                              int kt, int nt, int lane) {
  return *(const v16h*)(P + ((((size_t)kt * 16 + nt) * 32) + lane) * 16);
}

#define WMMA_F16(A, Bv, C) \
  __builtin_amdgcn_wmma_f32_16x16x32_f16(false, (A), false, (Bv), (short)0, (C), false, false)

// ---------------------------------------------------------------------------
// k_pack: convert W[K,256] fp32 -> f16 B-fragment-packed layout.
// One thread per (tile, lane): writes its 16 halves contiguously.
//   element (lane,slot s): v=s/2, j=s%2, kk=(v<4 ? v*2+j : 16+(v-4)*2+j)+(lane>>4)*8, n=lane&15
// ---------------------------------------------------------------------------
__global__ void k_pack(const float* __restrict__ W, int K,
                       _Float16* __restrict__ out) {
  const int idx   = blockIdx.x * blockDim.x + threadIdx.x;
  const int total = (K / 32) * 16 * 32;
  if (idx >= total) return;
  const int lane = idx & 31;
  const int tile = idx >> 5;
  const int nt   = tile & 15;
  const int kt   = tile >> 4;
  const int grp  = lane >> 4;
  const int n    = lane & 15;
  _Float16* dst = out + (size_t)idx * 16;
#pragma unroll
  for (int v = 0; v < 8; ++v) {
#pragma unroll
    for (int j = 0; j < 2; ++j) {
      int kk = (v < 4) ? (v * 2 + j + grp * 8) : (16 + (v - 4) * 2 + j + grp * 8);
      dst[v * 2 + j] = (_Float16)W[(size_t)(kt * 32 + kk) * HH + nt * 16 + n];
    }
  }
}

// ---------------------------------------------------------------------------
// K0: WvWo[e][o] = sum_a Wv[e,a] * Wo[a,o]   (32x32, one small block)
// ---------------------------------------------------------------------------
__global__ void k0_wvwo(const float* __restrict__ Wv, const float* __restrict__ Wo,
                        float* __restrict__ WvWo) {
  for (int i = threadIdx.x; i < EE * EE; i += blockDim.x) {
    int e = i / EE, o = i % EE;
    float acc = 0.f;
#pragma unroll 8
    for (int a = 0; a < AA; ++a) acc += Wv[e * AA + a] * Wo[a * EE + o];
    WvWo[i] = acc;
  }
}

// ---------------------------------------------------------------------------
// K1: per batch row -> tgt_e, qk = (Wk @ (tgt_e @ Wq)) / sqrt(A),
//     cat_pool, num_e.  Writes fused[b][0:32)=tgt, [64:96)=cat_pool,
//     [96:128)=num_e and qk[b][0:32).
// ---------------------------------------------------------------------------
__global__ void k1_prep(const int* __restrict__ cats, const float* __restrict__ nums,
                        const int* __restrict__ tgt_ids,
                        const float* __restrict__ cat_emb,
                        const float* __restrict__ Wnum, const float* __restrict__ bnum,
                        const float* __restrict__ Wq,   const float* __restrict__ Wk,
                        const float* __restrict__ Wpool,const float* __restrict__ bpool,
                        float* __restrict__ fused, float* __restrict__ qk) {
  __shared__ float s_tgt[EE];
  __shared__ float s_q[AA];
  __shared__ float s_cat[NCAT * EE];
  const int b   = blockIdx.x;
  const int tid = threadIdx.x;

  if (tid < EE) s_tgt[tid] = cat_emb[(size_t)tgt_ids[b] * EE + tid];
  for (int j = tid; j < NCAT * EE; j += blockDim.x) {
    int c = j / EE, e = j % EE;
    s_cat[j] = cat_emb[(size_t)cats[b * NCAT + c] * EE + e];
  }
  __syncthreads();

  // q = tgt_e @ Wq   (128 outputs, 1 per thread)
  {
    float acc = 0.f;
#pragma unroll
    for (int e = 0; e < EE; ++e) acc += s_tgt[e] * Wq[e * AA + tid];
    s_q[tid] = acc;
  }
  __syncthreads();

  if (tid < EE) {
    // qk = Wk @ q, scaled by 1/sqrt(A)
    float acc = 0.f;
#pragma unroll 8
    for (int a = 0; a < AA; ++a) acc += Wk[tid * AA + a] * s_q[a];
    qk[b * EE + tid] = acc * rsqrtf((float)AA);

    // cat_pool = cat_flat(640) @ Wpool + bpool
    float cp = bpool[tid];
    for (int j = 0; j < NCAT * EE; ++j) cp += s_cat[j] * Wpool[j * EE + tid];

    // num_e = nums @ Wnum + bnum
    float ne = bnum[tid];
#pragma unroll
    for (int j = 0; j < NNUM; ++j) ne += nums[b * NNUM + j] * Wnum[j * EE + tid];

    fused[b * 128 + tid]      = s_tgt[tid];
    fused[b * 128 + 64 + tid] = cp;
    fused[b * 128 + 96 + tid] = ne;
  }
}

// ---------------------------------------------------------------------------
// K2: per batch row -> 1024 gathered 32-dot scores, top-30 argmax, softmax,
//     weighted pooled embedding, interest = pooled @ WvWo + bo -> fused[32:64)
// ---------------------------------------------------------------------------
__global__ void k2_attn(const int* __restrict__ seqs, const float* __restrict__ seq_emb,
                        const float* __restrict__ bo,  const float* __restrict__ WvWo,
                        const float* __restrict__ qk,  float* __restrict__ fused) {
  __shared__ float s_sc[LL];
  __shared__ float s_rv[256];
  __shared__ int   s_ri[256];
  __shared__ float s_val[TOPK + 2];
  __shared__ int   s_idx[TOPK + 2];
  __shared__ float s_w[TOPK + 2];
  __shared__ float s_pool[EE];
  __shared__ float s_qkv[EE];

  const int b   = blockIdx.x;
  const int tid = threadIdx.x;

  if (tid < EE) s_qkv[tid] = qk[b * EE + tid];
  __syncthreads();

  // scores: random-index gathers hit the 6.4MB table resident in L2
  for (int l = tid; l < LL; l += 256) {
    if (l + 256 < LL)
      __builtin_prefetch(seq_emb + (size_t)seqs[b * LL + l + 256] * EE, 0, 0);
    const float* e = seq_emb + (size_t)seqs[b * LL + l] * EE;
    float s = 0.f;
#pragma unroll
    for (int j = 0; j < EE; ++j) s += e[j] * s_qkv[j];
    s_sc[l] = s;
  }
  __syncthreads();

  // iterative top-30 (argmax + invalidate), block tree reduction
  for (int k = 0; k < TOPK; ++k) {
    float bv = -1e30f;
    int   bi = 0;
    for (int l = tid; l < LL; l += 256) {
      float v = s_sc[l];
      if (v > bv) { bv = v; bi = l; }
    }
    s_rv[tid] = bv; s_ri[tid] = bi;
    __syncthreads();
    for (int st = 128; st > 0; st >>= 1) {
      if (tid < st && s_rv[tid + st] > s_rv[tid]) {
        s_rv[tid] = s_rv[tid + st]; s_ri[tid] = s_ri[tid + st];
      }
      __syncthreads();
    }
    if (tid == 0) {
      s_val[k] = s_rv[0]; s_idx[k] = s_ri[0];
      s_sc[s_ri[0]] = -1e30f;
    }
    __syncthreads();
  }

  // softmax over the 30 selected scores
  if (tid == 0) {
    float m = s_val[0];
    for (int k = 1; k < TOPK; ++k) m = fmaxf(m, s_val[k]);
    float sum = 0.f;
    for (int k = 0; k < TOPK; ++k) { float w = __expf(s_val[k] - m); s_w[k] = w; sum += w; }
    float inv = 1.f / sum;
    for (int k = 0; k < TOPK; ++k) s_w[k] *= inv;
  }
  __syncthreads();

  // pooled = sum_k w_k * seq_e[idx_k]  (32-dim)
  if (tid < EE) {
    float p = 0.f;
    for (int k = 0; k < TOPK; ++k)
      p += s_w[k] * seq_emb[(size_t)seqs[b * LL + s_idx[k]] * EE + tid];
    s_pool[tid] = p;
  }
  __syncthreads();

  // interest = pooled @ (Wv@Wo) + bo
  if (tid < EE) {
    float acc = bo[tid];
#pragma unroll
    for (int e = 0; e < EE; ++e) acc += s_pool[e] * WvWo[e * EE + tid];
    fused[b * 128 + 32 + tid] = acc;
  }
}

// ---------------------------------------------------------------------------
// K3: dense tail with WMMA. 16 batch rows per block, 128 threads (4 waves),
// each wave owns 4 N-tiles of the 256-wide activations. Activations in LDS;
// weights come pre-packed (f16, fragment layout, L2-resident). Fused tile
// staged LDS-side by the Tensor Data Mover.
//   h   = relu(fused[16,128] @ Wmlp + bmlp)
//   h   = qnn(h, q1_*) ; h = qnn(h, q2_*)
//   out = h @ Wout + bout
// ---------------------------------------------------------------------------
__global__ void k3_mlp(const float* __restrict__ fused,
                       const _Float16* __restrict__ Pmlp, const float* __restrict__ bmlp,
                       const _Float16* __restrict__ P11, const float* __restrict__ b11,
                       const _Float16* __restrict__ P12, const float* __restrict__ b12,
                       const _Float16* __restrict__ P1p, const float* __restrict__ b1p,
                       const _Float16* __restrict__ P21, const float* __restrict__ b21,
                       const _Float16* __restrict__ P22, const float* __restrict__ b22,
                       const _Float16* __restrict__ P2p, const float* __restrict__ b2p,
                       const float* __restrict__ Wout, const float* __restrict__ bout,
                       float* __restrict__ out) {
  __shared__ float    s_f32[16 * 128]; // TDM landing buffer (fused tile, f32)
  __shared__ _Float16 s_fh[16 * 128];  // fused tile, f16
  __shared__ float    s_h [16 * HH];   // current activations, f32
  __shared__ _Float16 s_hh[16 * HH];   // current activations, f16
  __shared__ float    s_t1[16 * HH];   // linear-1 / hnew
  __shared__ float    s_t2[16 * HH];   // linear-2
  __shared__ _Float16 s_qh[16 * HH];   // hadamard product, f16

  const int tid  = threadIdx.x;
  const int lane = tid & 31;
  const int wave = tid >> 5;
  const int b0   = blockIdx.x * 16;
  const int n    = lane & 15;
  const int grp  = lane >> 4;

  // ---- TDM: DMA the 16x128 fp32 fused tile (row stride 128) into LDS.
  // D# per CDNA5 ISA: group0 {count=1, lds_addr, global_addr, type=2},
  // group1 {data_size=4B, tensor_dim0=128, tensor_dim1=16, tile 128x16,
  // tensor_dim0_stride=128}. EXEC is ignored by TDM -> issue from wave 0 only.
  if (wave == 0) {
    const uint64_t ga = (uint64_t)(uintptr_t)(fused + (size_t)b0 * 128);
    const uint32_t la = (uint32_t)(uintptr_t)(&s_f32[0]); // LDS offset = low 32b
    u32x4 g0;
    g0[0] = 1u;                                            // count=1
    g0[1] = la;                                            // lds_addr
    g0[2] = (uint32_t)ga;                                  // global_addr[31:0]
    g0[3] = ((uint32_t)(ga >> 32) & 0x01FFFFFFu) | (2u << 30); // [56:32]+type=2
    u32x8 g1;
    g1[0] = 2u << 16;          // data_size = 4 bytes
    g1[1] = 128u << 16;        // tensor_dim0[15:0]=128 in bits[79:64... (31:16)]
    g1[2] = 16u  << 16;        // tensor_dim0 hi=0; tensor_dim1[15:0]=16
    g1[3] = 128u << 16;        // tensor_dim1 hi=0; tile_dim0=128
    g1[4] = 16u;               // tile_dim1=16; tile_dim2=0
    g1[5] = 128u;              // tensor_dim0_stride[31:0]=128
    g1[6] = 0u;                // stride hi / tensor_dim1_stride lo
    g1[7] = 0u;
    asm volatile("tensor_load_to_lds %0, %1" :: "s"(g0), "s"(g1) : "memory");
    __builtin_amdgcn_s_wait_tensorcnt(0);
  }
  __syncthreads();

  // convert fused tile to f16
  for (int i = tid; i < 16 * 128; i += 128) s_fh[i] = (_Float16)s_f32[i];
  __syncthreads();

  // ---- layer 1: h = relu(fused @ Wmlp + bmlp), K=128 -> 4 WMMA k-steps
  for (int t = 0; t < 4; ++t) {
    const int nt = wave * 4 + t;
    v8f acc = {};
#pragma unroll
    for (int kt = 0; kt < 4; ++kt) {
      v16h a  = load_a_frag(s_fh, 128, kt * 32, lane);
      v16h bb = load_b_packed(Pmlp, kt, nt, lane);
      acc = WMMA_F16(a, bb, acc);
    }
#pragma unroll
    for (int v = 0; v < 8; ++v) {
      int m = v + grp * 8, col = nt * 16 + n;
      float x = acc[v] + bmlp[col];
      s_h[m * HH + col] = x > 0.f ? x : 0.f;
    }
  }
  __syncthreads();
  for (int i = tid; i < 16 * HH; i += 128) s_hh[i] = (_Float16)s_h[i];
  __syncthreads();

  // ---- quadratic block: h <- h + ((h@W1+b1)*(h@W2+b2))@Wp + bp
  auto qnn = [&](const _Float16* P1, const float* b1,
                 const _Float16* P2, const float* b2,
                 const _Float16* Pp, const float* bp) {
    for (int t = 0; t < 4; ++t) {
      const int nt = wave * 4 + t;
      v8f a1 = {}, a2 = {};
#pragma unroll
      for (int kt = 0; kt < 8; ++kt) {
        v16h af  = load_a_frag(s_hh, HH, kt * 32, lane);
        v16h bf1 = load_b_packed(P1, kt, nt, lane);
        a1 = WMMA_F16(af, bf1, a1);
        v16h bf2 = load_b_packed(P2, kt, nt, lane);
        a2 = WMMA_F16(af, bf2, a2);
      }
#pragma unroll
      for (int v = 0; v < 8; ++v) {
        int m = v + grp * 8, col = nt * 16 + n;
        s_t1[m * HH + col] = a1[v] + b1[col];
        s_t2[m * HH + col] = a2[v] + b2[col];
      }
    }
    __syncthreads();
    for (int i = tid; i < 16 * HH; i += 128)
      s_qh[i] = (_Float16)(s_t1[i] * s_t2[i]);
    __syncthreads();
    for (int t = 0; t < 4; ++t) {
      const int nt = wave * 4 + t;
      v8f acc = {};
#pragma unroll
      for (int kt = 0; kt < 8; ++kt) {
        v16h af = load_a_frag(s_qh, HH, kt * 32, lane);
        v16h bf = load_b_packed(Pp, kt, nt, lane);
        acc = WMMA_F16(af, bf, acc);
      }
#pragma unroll
      for (int v = 0; v < 8; ++v) {
        int m = v + grp * 8, col = nt * 16 + n;
        s_t1[m * HH + col] = s_h[m * HH + col] + acc[v] + bp[col];
      }
    }
    __syncthreads();
    for (int i = tid; i < 16 * HH; i += 128) {
      float x = s_t1[i];
      s_h[i]  = x;
      s_hh[i] = (_Float16)x;
    }
    __syncthreads();
  };

  qnn(P11, b11, P12, b12, P1p, b1p);
  qnn(P21, b21, P22, b22, P2p, b2p);

  // ---- out = h @ Wout + bout  (one row per thread 0..15)
  if (tid < 16) {
    float acc = bout[0];
#pragma unroll 8
    for (int j = 0; j < HH; ++j) acc += s_h[tid * HH + j] * Wout[j];
    out[b0 + tid] = acc;
  }
}

// ---------------------------------------------------------------------------
extern "C" void kernel_launch(void* const* d_in, const int* in_sizes, int n_in,
                              void* d_out, int out_size, void* d_ws, size_t ws_size,
                              hipStream_t stream) {
  (void)in_sizes; (void)n_in; (void)out_size; (void)ws_size;
  const int*   cats    = (const int*)  d_in[0];
  const float* nums    = (const float*)d_in[1];
  const int*   seqs    = (const int*)  d_in[2];
  const int*   tgt_ids = (const int*)  d_in[3];
  const float* cat_emb = (const float*)d_in[4];
  const float* seq_emb = (const float*)d_in[5];
  const float* Wnum  = (const float*)d_in[6];
  const float* bnum  = (const float*)d_in[7];
  const float* Wq    = (const float*)d_in[8];
  const float* Wk    = (const float*)d_in[9];
  const float* Wv    = (const float*)d_in[10];
  const float* Wo    = (const float*)d_in[11];
  const float* bo    = (const float*)d_in[12];
  const float* Wpool = (const float*)d_in[13];
  const float* bpool = (const float*)d_in[14];
  const float* Wmlp  = (const float*)d_in[15];
  const float* bmlp  = (const float*)d_in[16];
  const float* q1W1  = (const float*)d_in[17];
  const float* q1b1  = (const float*)d_in[18];
  const float* q1W2  = (const float*)d_in[19];
  const float* q1b2  = (const float*)d_in[20];
  const float* q1Wp  = (const float*)d_in[21];
  const float* q1bp  = (const float*)d_in[22];
  const float* q2W1  = (const float*)d_in[23];
  const float* q2b1  = (const float*)d_in[24];
  const float* q2W2  = (const float*)d_in[25];
  const float* q2b2  = (const float*)d_in[26];
  const float* q2Wp  = (const float*)d_in[27];
  const float* q2bp  = (const float*)d_in[28];
  const float* Wout  = (const float*)d_in[29];
  const float* bout  = (const float*)d_in[30];

  float* ws    = (float*)d_ws;
  float* fused = ws;                  // B*128 f32
  float* qk    = fused + BB * 128;    // B*32  f32
  float* WvWo  = qk + BB * EE;        // 32*32 f32

  // packed f16 weights (B-fragment layout), 32B-aligned tail of workspace
  _Float16* wpk  = (_Float16*)(WvWo + EE * EE);
  _Float16* Pmlp = wpk;               // 128*256
  _Float16* P11  = Pmlp + 128 * HH;   // 256*256 each from here
  _Float16* P12  = P11 + HH * HH;
  _Float16* P1p  = P12 + HH * HH;
  _Float16* P21  = P1p + HH * HH;
  _Float16* P22  = P21 + HH * HH;
  _Float16* P2p  = P22 + HH * HH;

  // one-time-per-launch weight packing (deterministic, stream-ordered)
  k_pack<<<(4 * 16 * 32 + 255) / 256, 256, 0, stream>>>(Wmlp, 128, Pmlp);
  k_pack<<<(8 * 16 * 32 + 255) / 256, 256, 0, stream>>>(q1W1, 256, P11);
  k_pack<<<(8 * 16 * 32 + 255) / 256, 256, 0, stream>>>(q1W2, 256, P12);
  k_pack<<<(8 * 16 * 32 + 255) / 256, 256, 0, stream>>>(q1Wp, 256, P1p);
  k_pack<<<(8 * 16 * 32 + 255) / 256, 256, 0, stream>>>(q2W1, 256, P21);
  k_pack<<<(8 * 16 * 32 + 255) / 256, 256, 0, stream>>>(q2W2, 256, P22);
  k_pack<<<(8 * 16 * 32 + 255) / 256, 256, 0, stream>>>(q2Wp, 256, P2p);

  k0_wvwo<<<1, 256, 0, stream>>>(Wv, Wo, WvWo);
  k1_prep<<<BB, 128, 0, stream>>>(cats, nums, tgt_ids, cat_emb, Wnum, bnum,
                                  Wq, Wk, Wpool, bpool, fused, qk);
  k2_attn<<<BB, 256, 0, stream>>>(seqs, seq_emb, bo, WvWo, qk, fused);
  k3_mlp<<<BB / 16, 128, 0, stream>>>(fused, Pmlp, bmlp,
                                      P11, q1b1, P12, q1b2, P1p, q1bp,
                                      P21, q2b1, P22, q2b2, P2p, q2bp,
                                      Wout, bout, (float*)d_out);
}
